// GCNSPnet_90520730731083
// MI455X (gfx1250) — compile-verified, matrix operand order checked by reading, not run on hardware
//
#include <hip/hip_runtime.h>
#include <hip/hip_bf16.h>
#include <math.h>

// ---------------------------------------------------------------------------
// Types for WMMA intrinsics (gfx1250, wave32)
// ---------------------------------------------------------------------------
typedef __attribute__((ext_vector_type(16))) __bf16 v16bf;
typedef __attribute__((ext_vector_type(8)))  __bf16 v8bf;
typedef __attribute__((ext_vector_type(8)))  float  v8f;
typedef __attribute__((ext_vector_type(4)))  float  v4f;

#define B_SZ    256
#define N_NODE  400
#define F_IN    400
#define HID     64
#define PRED_IN 192   // 2*HID + E

#define KMAX_PAD 416              // ceil(400/32)*32
#define BSTR     (KMAX_PAD + 8)   // LDS row stride (elements), keeps 16B align

// ---------------------------------------------------------------------------
// Utility kernels
// ---------------------------------------------------------------------------
__global__ void k_zero(float* __restrict__ p, int n) {
  int i = blockIdx.x * blockDim.x + threadIdx.x;
  if (i < n) p[i] = 0.0f;
}

__global__ void k_scatter_adj(const int* __restrict__ rows,
                              const int* __restrict__ cols,
                              const float* __restrict__ vals,
                              float* __restrict__ adj, int nnz) {
  int i = blockIdx.x * blockDim.x + threadIdx.x;
  if (i < nnz) atomicAdd(adj + (size_t)rows[i] * N_NODE + cols[i], vals[i]);
}

// ---------------------------------------------------------------------------
// B fragment fetch from transposed bf16 LDS panel (layout == WMMA K-map)
// ---------------------------------------------------------------------------
__device__ __forceinline__ v16bf ldB(const __bf16* __restrict__ BT,
                                     int nt, int lm, int hb, int kk) {
  const __bf16* bp = BT + (nt * 16 + lm) * BSTR + kk + hb;
  v8bf blo = *(const v8bf*)bp;           // K = kk+hb+0..7
  v8bf bhi = *(const v8bf*)(bp + 16);    // K = kk+16+hb+0..7
  v16bf r;
  #pragma unroll
  for (int i = 0; i < 8; ++i) { r[i] = blo[i]; r[i + 8] = bhi[i]; }
  return r;
}

// ---------------------------------------------------------------------------
// bf16-WMMA GEMM specialized for N = 64:  D[M,64] = A[M,K] * B[K,64]
// fp32 in/out, bf16 compute, f32 accumulate (v_wmma_f32_16x16x32_bf16).
//
// Block = 256 threads (8 waves) -> 128x64 output tile.
// Whole B panel (K x 64) staged ONCE into LDS, transposed, bf16, zero-padded
// to a multiple of 32 in K (<= 52 KB).  Branch-free main k-loop:
//   A: 4x global_load_b128 clause + v_cvt_pk_bf16_f32
//   B: 8x ds_load_b128 issued up front (overlappable waits)
//   4 WMMAs per k-step; single vectorized tail step (B zero-padded).
// Batched via blockIdx.z with element strides sA/sB/sD.
// ---------------------------------------------------------------------------
__global__ __launch_bounds__(256)
void k_wmma_gemm_n64(const float* __restrict__ A, const float* __restrict__ B,
                     float* __restrict__ D,
                     int M, int K, int lda,
                     long long sA, long long sB, long long sD) {
  __shared__ __align__(16) __bf16 BT[64 * BSTR];   // BT[col][k]

  const int z = blockIdx.z;
  A += (size_t)z * sA;
  B += (size_t)z * sB;
  D += (size_t)z * sD;

  const int t    = threadIdx.x;
  const int Kpad = (K + 31) & ~31;

  // ---- stage B^T into LDS as bf16 (pairs of k -> one 32-bit LDS store) ----
  const int npairs = (Kpad >> 1) * 64;
  for (int e = t; e < npairs; e += 256) {
    int col = e & 63;
    int k   = (e >> 6) << 1;
    float v0 = (k     < K) ? B[(size_t)k * 64 + col]       : 0.0f;
    float v1 = (k + 1 < K) ? B[(size_t)(k + 1) * 64 + col] : 0.0f;
    union { __bf16 h[2]; unsigned int u; } p;
    p.h[0] = (__bf16)v0;
    p.h[1] = (__bf16)v1;
    *(unsigned int*)&BT[col * BSTR + k] = p.u;
  }
  __syncthreads();

  // ---- per-wave tile ----
  const int lane  = t & 31;
  const int wave  = t >> 5;
  const int mBase = blockIdx.x * 128 + wave * 16;
  if (mBase >= M) return;           // wave-uniform: EXEC all-1s inside

  const int lm = lane & 15;         // row (A) / col (B) within 16-tile
  const int hb = (lane >> 4) << 3;  // 0 (lanes 0-15) or 8 (lanes 16-31)

  v8f c0 = {0.f,0.f,0.f,0.f,0.f,0.f,0.f,0.f};
  v8f c1 = c0, c2 = c0, c3 = c0;

  const float* __restrict__ arow = A + (size_t)(mBase + lm) * lda;
  const int Kmain = K & ~31;
  const int rem   = K - Kmain;      // 0 or 16 at all call sites

  // ================= branch-free main k-loop =================
  for (int kk = 0; kk < Kmain; kk += 32) {
    // A fragment: lane holds K = kk+hb+0..7 and kk+16+hb+0..7
    v4f a0 = *(const v4f*)(arow + kk + hb);
    v4f a1 = *(const v4f*)(arow + kk + hb + 4);
    v4f a2 = *(const v4f*)(arow + kk + 16 + hb);
    v4f a3 = *(const v4f*)(arow + kk + 16 + hb + 4);
    __builtin_prefetch(arow + kk + 32 + hb, 0, 1);   // next A chunk -> cache
    v16bf af;
    #pragma unroll
    for (int j = 0; j < 4; ++j) {
      af[j]      = (__bf16)a0[j];
      af[j + 4]  = (__bf16)a1[j];
      af[j + 8]  = (__bf16)a2[j];
      af[j + 12] = (__bf16)a3[j];
    }
    // issue ALL B fragment DS loads before any WMMA (overlappable waits)
    v16bf b0 = ldB(BT, 0, lm, hb, kk);
    v16bf b1 = ldB(BT, 1, lm, hb, kk);
    v16bf b2 = ldB(BT, 2, lm, hb, kk);
    v16bf b3 = ldB(BT, 3, lm, hb, kk);
    // (neg_a, A, neg_b, B, c_mod, C, reuse_a, reuse_b)
    c0 = __builtin_amdgcn_wmma_f32_16x16x32_bf16(false, af, false, b0, (short)0, c0, false, false);
    c1 = __builtin_amdgcn_wmma_f32_16x16x32_bf16(false, af, false, b1, (short)0, c1, false, false);
    c2 = __builtin_amdgcn_wmma_f32_16x16x32_bf16(false, af, false, b2, (short)0, c2, false, false);
    c3 = __builtin_amdgcn_wmma_f32_16x16x32_bf16(false, af, false, b3, (short)0, c3, false, false);
  }

  // ================= single tail step (K % 32 != 0) =================
  if (rem) {
    v16bf af;
    if (rem == 16) {
      // low half valid: k = Kmain+hb+0..7 <= K-1 (hb<=8); high half hits
      // zero-padded B -> contributes nothing, load as zeros.
      v4f a0 = *(const v4f*)(arow + Kmain + hb);
      v4f a1 = *(const v4f*)(arow + Kmain + hb + 4);
      #pragma unroll
      for (int j = 0; j < 4; ++j) {
        af[j]      = (__bf16)a0[j];
        af[j + 4]  = (__bf16)a1[j];
        af[j + 8]  = (__bf16)0.0f;
        af[j + 12] = (__bf16)0.0f;
      }
    } else {
      #pragma unroll
      for (int i = 0; i < 16; ++i) {
        int k = Kmain + (i & 7) + ((i >> 3) << 4) + hb;
        af[i] = (k < K) ? (__bf16)arow[k] : (__bf16)0.0f;
      }
    }
    v16bf b0 = ldB(BT, 0, lm, hb, Kmain);
    v16bf b1 = ldB(BT, 1, lm, hb, Kmain);
    v16bf b2 = ldB(BT, 2, lm, hb, Kmain);
    v16bf b3 = ldB(BT, 3, lm, hb, Kmain);
    c0 = __builtin_amdgcn_wmma_f32_16x16x32_bf16(false, af, false, b0, (short)0, c0, false, false);
    c1 = __builtin_amdgcn_wmma_f32_16x16x32_bf16(false, af, false, b1, (short)0, c1, false, false);
    c2 = __builtin_amdgcn_wmma_f32_16x16x32_bf16(false, af, false, b2, (short)0, c2, false, false);
    c3 = __builtin_amdgcn_wmma_f32_16x16x32_bf16(false, af, false, b3, (short)0, c3, false, false);
  }

  // ---- store: VGPR r -> row r (lanes 0-15) or r+8 (lanes 16-31) ----
  const int row0 = mBase + hb;
  #pragma unroll
  for (int r = 0; r < 8; ++r) {
    float* drow = D + (size_t)(row0 + r) * 64 + lm;
    drow[0]  = c0[r];
    drow[16] = c1[r];
    drow[32] = c2[r];
    drow[48] = c3[r];
  }
}

// ---------------------------------------------------------------------------
// y = g + bias;  y /= max(||y||_2, 1e-12)  (per row of 64);  optional ReLU
// grid = B*N rows, block = 64
// ---------------------------------------------------------------------------
__global__ void k_bias_l2norm(const float* __restrict__ g,
                              const float* __restrict__ bias,
                              float* __restrict__ out, int relu) {
  __shared__ float s[64];
  const int row = blockIdx.x;
  const int f   = threadIdx.x;
  float y = g[(size_t)row * HID + f] + bias[f];
  s[f] = y * y;
  __syncthreads();
  for (int st = 32; st > 0; st >>= 1) {
    if (f < st) s[f] += s[f + st];
    __syncthreads();
  }
  float inv = 1.0f / fmaxf(sqrtf(s[0]), 1e-12f);
  y *= inv;
  if (relu) y = fmaxf(y, 0.0f);
  out[(size_t)row * HID + f] = y;
}

// ---------------------------------------------------------------------------
// BatchNorm over node-channel dim: stats over (B, H) for each n in [0,400)
// grid = N_NODE, block = 256
// ---------------------------------------------------------------------------
__global__ void k_bn3_stats(const float* __restrict__ x,
                            float* __restrict__ mean, float* __restrict__ rstd) {
  __shared__ float ss[256], ss2[256];
  const int n = blockIdx.x;
  const int t = threadIdx.x;
  const int f  = t & 63;
  const int b0 = t >> 6;
  float s = 0.f, s2 = 0.f;
  for (int b = b0; b < B_SZ; b += 4) {
    float v = x[((size_t)b * N_NODE + n) * HID + f];
    s += v; s2 += v * v;
  }
  ss[t] = s; ss2[t] = s2;
  __syncthreads();
  for (int st = 128; st > 0; st >>= 1) {
    if (t < st) { ss[t] += ss[t + st]; ss2[t] += ss2[t + st]; }
    __syncthreads();
  }
  if (t == 0) {
    const float inv = 1.0f / (float)(B_SZ * HID);
    float m = ss[0] * inv;
    float v = ss2[0] * inv - m * m;
    mean[n] = m;
    rstd[n] = rsqrtf(v + 1e-5f);
  }
}

// apply bn + max over n, writing max into h[b, colOff + f] (h row stride 192)
// grid = B_SZ, block = (64, 8)
__global__ void k_bn3_apply_max(const float* __restrict__ x,
                                const float* __restrict__ mean,
                                const float* __restrict__ rstd,
                                float* __restrict__ xbn,
                                float* __restrict__ hbuf, int colOff) {
  __shared__ float red[8][64];
  const int b = blockIdx.x;
  const int f = threadIdx.x;
  const int j = threadIdx.y;
  float mx = -3.4e38f;
  for (int n = j; n < N_NODE; n += 8) {
    size_t idx = ((size_t)b * N_NODE + n) * HID + f;
    float v = (x[idx] - mean[n]) * rstd[n];
    xbn[idx] = v;
    mx = fmaxf(mx, v);
  }
  red[j][f] = mx;
  __syncthreads();
  if (j == 0) {
    float m = red[0][f];
    #pragma unroll
    for (int q = 1; q < 8; ++q) m = fmaxf(m, red[q][f]);
    hbuf[(size_t)b * PRED_IN + colOff + f] = m;
  }
}

// plain max over n (layer 3, no bn)
__global__ void k_max_over_n(const float* __restrict__ x,
                             float* __restrict__ hbuf, int colOff) {
  __shared__ float red[8][64];
  const int b = blockIdx.x;
  const int f = threadIdx.x;
  const int j = threadIdx.y;
  float mx = -3.4e38f;
  for (int n = j; n < N_NODE; n += 8)
    mx = fmaxf(mx, x[((size_t)b * N_NODE + n) * HID + f]);
  red[j][f] = mx;
  __syncthreads();
  if (j == 0) {
    float m = red[0][f];
    #pragma unroll
    for (int q = 1; q < 8; ++q) m = fmaxf(m, red[q][f]);
    hbuf[(size_t)b * PRED_IN + colOff + f] = m;
  }
}

// ---------------------------------------------------------------------------
// tiny MLP head
// ---------------------------------------------------------------------------
__global__ void k_fc(const float* __restrict__ hin, const float* __restrict__ W,
                     const float* __restrict__ bias, float* __restrict__ hout,
                     int IN, int OUT, int relu) {
  const int b = blockIdx.x, o = threadIdx.x;
  if (o >= OUT) return;
  float acc = bias[o];
  for (int i = 0; i < IN; ++i)
    acc += hin[(size_t)b * IN + i] * W[(size_t)i * OUT + o];
  if (relu) acc = fmaxf(acc, 0.0f);
  hout[(size_t)b * OUT + o] = acc;
}

// BatchNorm over batch dim, per column c; in-place. grid = C, block = 256
__global__ void k_bn2(float* __restrict__ h, int C) {
  __shared__ float ss[256], ss2[256];
  const int c = blockIdx.x, t = threadIdx.x;
  float v = h[(size_t)t * C + c];
  ss[t] = v; ss2[t] = v * v;
  __syncthreads();
  for (int st = 128; st > 0; st >>= 1) {
    if (t < st) { ss[t] += ss[t + st]; ss2[t] += ss2[t + st]; }
    __syncthreads();
  }
  float m   = ss[0] * (1.0f / 256.0f);
  float var = ss2[0] * (1.0f / 256.0f) - m * m;
  h[(size_t)t * C + c] = (v - m) * rsqrtf(var + 1e-5f);
}

// ---------------------------------------------------------------------------
// host orchestration
// ---------------------------------------------------------------------------
extern "C" void kernel_launch(void* const* d_in, const int* in_sizes, int n_in,
                              void* d_out, int out_size, void* d_ws, size_t ws_size,
                              hipStream_t stream) {
  const float* x        = (const float*)d_in[0];
  const int*   adj_rows = (const int*)  d_in[1];
  const int*   adj_cols = (const int*)  d_in[2];
  const float* adj_vals = (const float*)d_in[3];
  const float* W1 = (const float*)d_in[4];  const float* b1 = (const float*)d_in[5];
  const float* W2 = (const float*)d_in[6];  const float* b2 = (const float*)d_in[7];
  const float* W3 = (const float*)d_in[8];  const float* b3 = (const float*)d_in[9];
  const float* fc1W = (const float*)d_in[10]; const float* fc1b = (const float*)d_in[11];
  const float* fc2W = (const float*)d_in[12]; const float* fc2b = (const float*)d_in[13];
  const float* fc3W = (const float*)d_in[14]; const float* fc3b = (const float*)d_in[15];

  float* out   = (float*)d_out;          // [0,512): ypred ; [512,...): x3
  float* x3out = out + 512;

  // workspace layout (floats)
  float* ws   = (float*)d_ws;
  const long long BNH = (long long)B_SZ * N_NODE * HID;   // 6,553,600
  float* adj  = ws;                       // 160,000
  float* zb   = adj + (N_NODE * N_NODE);  // ping
  float* gb   = zb + BNH;                 // pong
  float* stat = gb + BNH;                 // mean[400] + rstd[400]
  float* hbuf = stat + 2 * N_NODE;        // [256, 192]
  float* h1   = hbuf + B_SZ * PRED_IN;    // [256, 64]
  float* h2   = h1 + B_SZ * HID;          // [256, 64]
  (void)n_in; (void)in_sizes; (void)out_size; (void)ws_size;

  const int NNZ = in_sizes[1];
  const int Mx  = B_SZ * N_NODE;          // 102,400 rows
  const long long sBD = (long long)N_NODE * HID;   // 25,600

  // 0) densify adjacency (re-zero each call: deterministic / replay-safe)
  k_zero<<<(N_NODE * N_NODE + 255) / 256, 256, 0, stream>>>(adj, N_NODE * N_NODE);
  k_scatter_adj<<<(NNZ + 255) / 256, 256, 0, stream>>>(adj_rows, adj_cols, adj_vals, adj, NNZ);

  dim3 gBig(Mx / 128, 1, 1);                       // (800, 1, 1)
  dim3 gAgg((N_NODE + 127) / 128, 1, B_SZ);        // (4, 1, 256)
  dim3 bApply(64, 8);

  // ---- layer 1:  z = X@W1 ; g = adj@z ; l2+relu ; bn3 ; max -> h[:,0:64]
  k_wmma_gemm_n64<<<gBig, 256, 0, stream>>>(x, W1, zb, Mx, F_IN, F_IN, 0, 0, 0);
  k_wmma_gemm_n64<<<gAgg, 256, 0, stream>>>(adj, zb, gb, N_NODE, N_NODE, N_NODE, 0, sBD, sBD);
  k_bias_l2norm<<<Mx, 64, 0, stream>>>(gb, b1, gb, 1);
  k_bn3_stats<<<N_NODE, 256, 0, stream>>>(gb, stat, stat + N_NODE);
  k_bn3_apply_max<<<B_SZ, bApply, 0, stream>>>(gb, stat, stat + N_NODE, zb, hbuf, 0);
  // zb now holds x1bn

  // ---- layer 2
  k_wmma_gemm_n64<<<gBig, 256, 0, stream>>>(zb, W2, gb, Mx, HID, HID, 0, 0, 0);
  k_wmma_gemm_n64<<<gAgg, 256, 0, stream>>>(adj, gb, zb, N_NODE, N_NODE, N_NODE, 0, sBD, sBD);
  k_bias_l2norm<<<Mx, 64, 0, stream>>>(zb, b2, zb, 1);
  k_bn3_stats<<<N_NODE, 256, 0, stream>>>(zb, stat, stat + N_NODE);
  k_bn3_apply_max<<<B_SZ, bApply, 0, stream>>>(zb, stat, stat + N_NODE, gb, hbuf, HID);
  // gb now holds x2bn

  // ---- layer 3 (no relu/bn); x3 straight into d_out
  k_wmma_gemm_n64<<<gBig, 256, 0, stream>>>(gb, W3, zb, Mx, HID, HID, 0, 0, 0);
  k_wmma_gemm_n64<<<gAgg, 256, 0, stream>>>(adj, zb, gb, N_NODE, N_NODE, N_NODE, 0, sBD, sBD);
  k_bias_l2norm<<<Mx, 64, 0, stream>>>(gb, b3, x3out, 0);
  k_max_over_n<<<B_SZ, bApply, 0, stream>>>(x3out, hbuf, 2 * HID);

  // ---- MLP head
  k_fc<<<B_SZ, 64, 0, stream>>>(hbuf, fc1W, fc1b, h1, PRED_IN, HID, 1);
  k_bn2<<<HID, 256, 0, stream>>>(h1, HID);
  k_fc<<<B_SZ, 64, 0, stream>>>(h1, fc2W, fc2b, h2, HID, HID, 1);
  k_bn2<<<HID, 256, 0, stream>>>(h2, HID);
  k_fc<<<B_SZ, 64, 0, stream>>>(h2, fc3W, fc3b, out, HID, 2, 0);
}
